// SerialAttention_78460462563653
// MI455X (gfx1250) — matrix operand order repaired and checked by reading, not compiled, not run on hardware
//
#include <hip/hip_runtime.h>
#include <hip/hip_bf16.h>
#include <stdint.h>

// ---------------------------------------------------------------------------
// Types
// ---------------------------------------------------------------------------
typedef __bf16 bf16;
typedef __attribute__((ext_vector_type(16))) __bf16 v16bf;
typedef __attribute__((ext_vector_type(8)))  float  v8f;

#define DEV __device__ __forceinline__

// Problem constants (match reference)
#define B_  4
#define S_  1024
#define D_  1024
#define H_  16
#define HKV_ 4
#define HD_ 64
#define QD_ 1024      // H*HD
#define KD_ 256       // HKV*HD
#define OH_ 64
#define G4D_ 4096     // 4*D
#define MTOT_ 4096    // B*S

// ---------------------------------------------------------------------------
// CDNA5 feature probes (device pass only)
// ---------------------------------------------------------------------------
#if defined(__gfx1250__) && __has_builtin(__builtin_amdgcn_tensor_load_to_lds) && \
    __has_builtin(__builtin_amdgcn_s_wait_tensorcnt)
#define HAVE_TDM 1
#else
#define HAVE_TDM 0
#endif

#if defined(__gfx1250__) && __has_builtin(__builtin_amdgcn_global_load_async_to_lds_b128)
#define HAVE_ASYNC 1
#else
#define HAVE_ASYNC 0
#endif

// ---------------------------------------------------------------------------
// WMMA helpers (CDNA5 gfx1250, wave32)
// ---------------------------------------------------------------------------
DEV v8f wmma_bf16(v16bf a, v16bf b, v8f c) {
  return __builtin_amdgcn_wmma_f32_16x16x32_bf16(
      /*neg_a=*/false, a, /*neg_b=*/false, b,
      /*c_mod=*/(short)0, c, /*reuse_a=*/false, /*reuse_b=*/false);
}

// A fragment: 16x32 bf16, row m = lane%16, k contiguous (matches frag_b):
//   VGPR j holds k = g*16 + j*2 (pairs), g = lane>>4
DEV v16bf frag_a(const uint32_t* rowbase, int ld_dw, int lane) {
  int m = lane & 15, g = lane >> 4;
  union { v16bf v; uint32_t u[8]; } f;
  const uint32_t* p = rowbase + (size_t)m * ld_dw;
#pragma unroll
  for (int j = 0; j < 8; ++j)
    f.u[j] = p[(g << 3) + j];
  return f.v;
}

// B fragment: 32x16 bf16 (stored as [n][k] row-major, n = lane%16):
//   lanes 0-15 hold k=0..15, lanes 16-31 hold k=16..31 -> dword index g*8 + j
DEV v16bf frag_b(const uint32_t* rowbase, int ld_dw, int lane) {
  int n = lane & 15, g = lane >> 4;
  union { v16bf v; uint32_t u[8]; } f;
  const uint32_t* p = rowbase + (size_t)n * ld_dw + (g << 3);
#pragma unroll
  for (int j = 0; j < 8; ++j) f.u[j] = p[j];
  return f.v;
}

DEV void wave_lds_fence() {
  asm volatile("s_wait_dscnt 0x0" ::: "memory");
}

// ---------------------------------------------------------------------------
// TDM tile load: 64 rows x 32 bf16 from row-major [*, row_elems] into LDS.
// D# built per CDNA5 ISA ch.8 (group0 128b, group1 256b; groups 2/3 zero).
// Toolchain builtin is the 6-arg form: (u32x4, i32x8, i32x4, i32x4, i32x8, cpol)
// ---------------------------------------------------------------------------
#if HAVE_TDM
typedef __attribute__((ext_vector_type(4))) unsigned u32x4;
typedef __attribute__((ext_vector_type(8))) int i32x8;
typedef __attribute__((ext_vector_type(4))) int i32x4;

typedef __attribute__((address_space(3))) void lds_void_t;

DEV unsigned lds_off(void* p) {
  return (unsigned)(uintptr_t)(lds_void_t*)p;
}

DEV void tdm_load_tile_64x32_bf16(unsigned lds_byte, const void* gsrc,
                                  unsigned row_elems) {
  unsigned long long ga = (unsigned long long)(uintptr_t)gsrc;
  u32x4 g0;
  g0.x = 1u;                                  // count=1, user mode, no gather
  g0.y = lds_byte;                            // lds_addr (bytes)
  g0.z = (unsigned)ga;                        // global_addr[31:0]
  g0.w = ((unsigned)(ga >> 32) & 0x01FFFFFFu) // global_addr[56:32]
         | (2u << 30);                        // type = 2 ("image")
  unsigned tdim0 = row_elems;                 // tensor width  (tile in-bounds)
  unsigned tdim1 = 1u << 20;                  // tensor height (tile in-bounds)
  unsigned strd  = row_elems;                 // dim0 stride (data_size units)
  i32x8 g1;
  g1[0] = (int)(1u << 16);                    // wg_mask=0, data_size=1 (2B)
  g1[1] = (int)((tdim0 & 0xFFFFu) << 16);     // abar=0 | tensor_dim0[15:0]
  g1[2] = (int)(((tdim0 >> 16) & 0xFFFFu) | ((tdim1 & 0xFFFFu) << 16));
  g1[3] = (int)(((tdim1 >> 16) & 0xFFFFu) | (32u << 16));  // tile_dim0 = 32
  g1[4] = (int)64;                            // tile_dim1 = 64, tile_dim2 = 0
  g1[5] = (int)strd;                          // dim0_stride[31:0]
  g1[6] = 0;                                  // dim0_stride[47:32] | d1strd lo
  g1[7] = 0;
  i32x4 gz4 = {0, 0, 0, 0};
  i32x8 gz8 = {0, 0, 0, 0, 0, 0, 0, 0};
  __builtin_amdgcn_tensor_load_to_lds(g0, g1, gz4, gz4, gz8, 0);
}
#endif  // HAVE_TDM

// ---------------------------------------------------------------------------
// Async global->LDS helpers for the attention K tile
// (builtin takes pointers to 128-bit int vectors: global src, LDS dst)
// ---------------------------------------------------------------------------
#if HAVE_ASYNC
typedef __attribute__((ext_vector_type(4))) int v4i_t;
typedef __attribute__((address_space(1))) v4i_t g_v4i_t;
typedef __attribute__((address_space(3))) v4i_t l_v4i_t;

DEV void async_ldst_wait() {
#if __has_builtin(__builtin_amdgcn_s_wait_asynccnt)
  __builtin_amdgcn_s_wait_asynccnt(0);
#else
  asm volatile("s_wait_asynccnt 0x0" ::: "memory");
#endif
}
#endif  // HAVE_ASYNC

// ---------------------------------------------------------------------------
// f32 -> bf16 cast kernel
// ---------------------------------------------------------------------------
__global__ void cast_f32_bf16_kernel(const float* __restrict__ in,
                                     bf16* __restrict__ out, int n) {
  int i = blockIdx.x * blockDim.x + threadIdx.x;
  if (i < n) out[i] = (bf16)in[i];
}

// ---------------------------------------------------------------------------
// Generic GEMM: C[M,N] = A[M,K] * B[N,K]^T (+bias) (opt relu)
// A,B bf16 row-major. Block 256 thr = 8 waves; tile 64x64, K-chunk 32.
// Tiles are DMA'd into LDS by the Tensor Data Mover when available.
// ---------------------------------------------------------------------------
__global__ __launch_bounds__(256)
void gemm_wmma_kernel(const bf16* __restrict__ A, const bf16* __restrict__ Bw,
                      const float* __restrict__ bias,
                      float* __restrict__ outF, bf16* __restrict__ outB,
                      int M, int N, int K, int relu) {
  __shared__ uint32_t As[64 * 16];
  __shared__ uint32_t Bs[64 * 16];
  const int tid = threadIdx.x;
  const int wid = tid >> 5, lane = tid & 31;
  const int wm = wid >> 1, wn = wid & 1;
  const int bm0 = blockIdx.y * 64, bn0 = blockIdx.x * 64;
  const int Kdw = K >> 1;
  v8f acc0 = {}; v8f acc1 = {};

#if HAVE_TDM
  const unsigned asl = lds_off(As);
  const unsigned bsl = lds_off(Bs);
#endif

  for (int kc = 0; kc < K; kc += 32) {
    __syncthreads();
#if HAVE_TDM
    if (wid == 0) {
      tdm_load_tile_64x32_bf16(asl, (const char*)A + ((size_t)bm0 * K + kc) * 2,
                               (unsigned)K);
      tdm_load_tile_64x32_bf16(bsl, (const char*)Bw + ((size_t)bn0 * K + kc) * 2,
                               (unsigned)K);
      __builtin_amdgcn_s_wait_tensorcnt(0);
    }
#else
    {
      const uint32_t* Ag = (const uint32_t*)A + (size_t)bm0 * Kdw + (kc >> 1);
      const uint32_t* Bg = (const uint32_t*)Bw + (size_t)bn0 * Kdw + (kc >> 1);
#pragma unroll
      for (int i = tid; i < 64 * 16; i += 256) {
        int r = i >> 4, c = i & 15;
        As[i] = Ag[(size_t)r * Kdw + c];
        Bs[i] = Bg[(size_t)r * Kdw + c];
      }
    }
#endif
    // speculative prefetch of the next K-chunk (global_prefetch_b8)
    if (kc + 32 < K) {
      __builtin_prefetch((const char*)A + ((size_t)(bm0 + (tid >> 2)) * K + kc + 32) * 2, 0, 1);
      __builtin_prefetch((const char*)Bw + ((size_t)(bn0 + (tid >> 2)) * K + kc + 32) * 2, 0, 1);
    }
    __syncthreads();
    v16bf af = frag_a(As + (wm * 16) * 16, 16, lane);
    v16bf b0 = frag_b(Bs + (wn * 32) * 16, 16, lane);
    v16bf b1 = frag_b(Bs + (wn * 32 + 16) * 16, 16, lane);
    acc0 = wmma_bf16(af, b0, acc0);
    acc1 = wmma_bf16(af, b1, acc1);
  }

  const int g = lane >> 4, nl = lane & 15;
  const int row0 = bm0 + wm * 16 + g * 8;
  const int col0 = bn0 + wn * 32;
#pragma unroll
  for (int r = 0; r < 8; ++r) {
    int row = row0 + r;
    float v0 = acc0[r];
    float v1 = acc1[r];
    int c0 = col0 + nl, c1 = col0 + 16 + nl;
    if (bias) { v0 += bias[c0]; v1 += bias[c1]; }
    if (relu) { v0 = v0 > 0.f ? v0 : 0.f; v1 = v1 > 0.f ? v1 : 0.f; }
    if (outF) {
      outF[(size_t)row * N + c0] = v0;
      outF[(size_t)row * N + c1] = v1;
    }
    if (outB) {
      outB[(size_t)row * N + c0] = (bf16)v0;
      outB[(size_t)row * N + c1] = (bf16)v1;
    }
  }
}

// ---------------------------------------------------------------------------
// Grid-wide barrier (monotone generation counter, reset via memsetAsync)
// ---------------------------------------------------------------------------
DEV void grid_sync(unsigned* bar, unsigned nblocks) {
  __threadfence();
  __syncthreads();
  if (threadIdx.x == 0) {
    unsigned arrive = atomicAdd(bar, 1u) + 1u;
    unsigned target = ((arrive + nblocks - 1u) / nblocks) * nblocks;
    while (atomicAdd(bar, 0u) < target) { __builtin_amdgcn_s_sleep(2); }
  }
  __syncthreads();
  __threadfence();
}

// ---------------------------------------------------------------------------
// Persistent LSTM: gates[b,j] = xW[b,t,j] + sum_d h[b,d]*Whh[j,d]
// 32 blocks x 8 waves = 256 waves, one 16-col gate tile per wave.
// h padded to 16 rows (rows 4..15 = 0) so M=16 WMMA is exact.
// ---------------------------------------------------------------------------
__global__ __launch_bounds__(256)
void lstm_kernel(const float* __restrict__ xW,      // [B,S,4D] f32 (bias folded)
                 const bf16* __restrict__ Whh,      // [4D,D] bf16
                 bf16* __restrict__ h_seq,          // [B,S,D] bf16 out
                 bf16* h_pad,                       // [16,D] scratch
                 float* c_state,                    // [B,D]
                 float* gates,                      // [B,4D]
                 unsigned* bar) {
  const int tid = threadIdx.x;
  const int lane = tid & 31, wid = tid >> 5;
  const int gtid = blockIdx.x * 256 + tid;
  const unsigned nblocks = gridDim.x;
  const int n0 = (blockIdx.x * 8 + wid) * 16;     // gate column tile

  // init state
  for (int i = gtid; i < 16 * D_; i += nblocks * 256) h_pad[i] = (bf16)0.f;
  for (int i = gtid; i < B_ * D_;  i += nblocks * 256) c_state[i] = 0.f;
  grid_sync(bar, nblocks);

  const uint32_t* hp = (const uint32_t*)h_pad;              // [16][D/2] dwords
  const uint32_t* Wd = (const uint32_t*)Whh;                // [4D][D/2]

  for (int t = 0; t < S_; ++t) {
    // Phase A: recurrent GEMM tile (16x16, K=1024)
    v8f acc = {};
#pragma unroll 4
    for (int kc = 0; kc < D_; kc += 32) {
      v16bf a = frag_a(hp + (kc >> 1), D_ / 2, lane);
      v16bf b = frag_b(Wd + (size_t)n0 * (D_ / 2) + (kc >> 1), D_ / 2, lane);
      acc = wmma_bf16(a, b, acc);
    }
    if (lane < 16) {   // rows 0..3 are the real batch rows (g==0, r<4)
#pragma unroll
      for (int r = 0; r < 4; ++r) gates[r * G4D_ + n0 + lane] = acc[r];
    }
    // prefetch next step's xW slice while the barrier settles
    if (t + 1 < S_ && gtid < B_ * D_) {
      int bb = gtid >> 10;
      __builtin_prefetch(xW + ((size_t)(bb * S_ + t + 1)) * G4D_ + (gtid & (D_ - 1)), 0, 1);
    }
    grid_sync(bar, nblocks);

    // Phase B: pointwise cell update (4096 lanes of work)
    if (gtid < B_ * D_) {
      int bb = gtid >> 10, d = gtid & (D_ - 1);
      const float* xt = xW + ((size_t)(bb * S_ + t)) * G4D_;
      const float* gr = gates + bb * G4D_;
      float gi = gr[d]            + xt[d];
      float gf = gr[D_ + d]       + xt[D_ + d];
      float gg = gr[2 * D_ + d]   + xt[2 * D_ + d];
      float go = gr[3 * D_ + d]   + xt[3 * D_ + d];
      float si = 1.f / (1.f + __expf(-gi));
      float sf = 1.f / (1.f + __expf(-gf));
      float so = 1.f / (1.f + __expf(-go));
      float cn = sf * c_state[gtid] + si * tanhf(gg);
      float hh = so * tanhf(cn);
      c_state[gtid] = cn;
      h_pad[bb * D_ + d] = (bf16)hh;
      h_seq[((size_t)(bb * S_ + t)) * D_ + d] = (bf16)hh;
    }
    grid_sync(bar, nblocks);
  }
}

// ---------------------------------------------------------------------------
// Causal GQA flash attention with WMMA.
// Block = 128 thr (4 waves); wave owns a 16-query tile; keys in 32-chunks.
// K tile is filled with async global->LDS copies (ASYNCcnt) when available.
// grid = (S/(4*16), B*H)
// ---------------------------------------------------------------------------
__global__ __launch_bounds__(128)
void attn_kernel(const bf16* __restrict__ Q,   // [B,S,QD]
                 const bf16* __restrict__ Kin, // [B,S,KD]
                 const bf16* __restrict__ V,   // [B,S,KD]
                 bf16* __restrict__ O) {       // [B,S,QD]
  const int tid = threadIdx.x, lane = tid & 31, wid = tid >> 5;
  const int bh = blockIdx.y;
  const int bb = bh >> 4, h = bh & 15;
  const int hk = h >> 2;                       // GQA group of 4
  const int q0 = blockIdx.x * 64 + wid * 16;
  const float scale = 0.125f;                  // 1/sqrt(64)
  const int g = lane >> 4, nl = lane & 15;

  // Per-wave LDS: Ks 32x64 bf16 (1024 dw) | Vt 64x32 bf16 (1024 dw) | P 16x32 (256 dw)
  __shared__ uint32_t smem[4][2304];
  uint32_t* Ks = smem[wid];
  uint32_t* Vt = Ks + 1024;
  uint32_t* Pm = Vt + 1024;

  // Q fragments (d 0..31 and 32..63), loaded once
  const uint32_t* Qb =
      (const uint32_t*)Q + ((size_t)(bb * S_ + q0)) * (QD_ / 2) + h * (HD_ / 2);
  v16bf qa0 = frag_a(Qb, QD_ / 2, lane);
  v16bf qa1 = frag_a(Qb + 16, QD_ / 2, lane);

  v8f o0 = {}, o1 = {}, o2 = {}, o3 = {};
  float m_run[8], l_run[8];
#pragma unroll
  for (int e = 0; e < 8; ++e) { m_run[e] = -3.0e38f; l_run[e] = 0.f; }

  const int nch = (q0 + 16 + 31) >> 5;
  for (int kt = 0; kt < nch; ++kt) {
    const int c0 = kt * 32;
    // K tile: 32 keys x 64 dims (row per lane)
    const char* Kg = (const char*)Kin +
        (((size_t)(bb * S_ + c0)) * KD_ + hk * HD_) * 2;
#if HAVE_ASYNC
    {
      const char* grow = Kg + (size_t)lane * (KD_ * 2);
      char* lrow = (char*)Ks + lane * 64;
#pragma unroll
      for (int c = 0; c < 4; ++c)
        __builtin_amdgcn_global_load_async_to_lds_b128(
            (g_v4i_t*)(grow + c * 16), (l_v4i_t*)(lrow + c * 16), 0, 0);
      async_ldst_wait();
    }
#else
    {
      const uint32_t* Kg32 = (const uint32_t*)Kg;
      for (int i = lane; i < 1024; i += 32) {
        int kk = i >> 5, dw = i & 31;
        Ks[kk * 32 + dw] = Kg32[(size_t)kk * (KD_ / 2) + dw];
      }
    }
#endif
    // V tile, transposed into [d][k] so (k,k+1) pairs are contiguous
    const bf16* Vg = V + ((size_t)(bb * S_ + c0)) * KD_ + hk * HD_;
    bf16* VtB = (bf16*)Vt;
    for (int i = lane; i < 2048; i += 32) {
      int kk = i >> 6, d = i & 63;
      VtB[d * 32 + kk] = Vg[(size_t)kk * KD_ + d];
    }
    wave_lds_fence();

    // scores: two 16-key sub-tiles, each QK^T over d=64 (2 WMMAs)
    v8f s0 = {}, s1 = {};
    s0 = wmma_bf16(qa0, frag_b(Ks, 32, lane), s0);
    s0 = wmma_bf16(qa1, frag_b(Ks + 16, 32, lane), s0);
    s1 = wmma_bf16(qa0, frag_b(Ks + 16 * 32, 32, lane), s1);
    s1 = wmma_bf16(qa1, frag_b(Ks + 16 * 32 + 16, 32, lane), s1);

    float p0[8], p1[8];
#pragma unroll
    for (int e = 0; e < 8; ++e) {
      int row = q0 + e + 8 * g;
      float sa = s0[e] * scale;
      float sb = s1[e] * scale;
      if (c0 + nl > row)       sa = -3.0e38f;
      if (c0 + 16 + nl > row)  sb = -3.0e38f;
      float mloc = fmaxf(sa, sb);
      mloc = fmaxf(mloc, __shfl_xor(mloc, 1));
      mloc = fmaxf(mloc, __shfl_xor(mloc, 2));
      mloc = fmaxf(mloc, __shfl_xor(mloc, 4));
      mloc = fmaxf(mloc, __shfl_xor(mloc, 8));
      float mnew = fmaxf(m_run[e], mloc);
      float pa = __expf(sa - mnew);
      float pb = __expf(sb - mnew);
      float rs = pa + pb;
      rs += __shfl_xor(rs, 1); rs += __shfl_xor(rs, 2);
      rs += __shfl_xor(rs, 4); rs += __shfl_xor(rs, 8);
      float alpha = __expf(m_run[e] - mnew);
      l_run[e] = l_run[e] * alpha + rs;
      m_run[e] = mnew;
      o0[e] *= alpha; o1[e] *= alpha; o2[e] *= alpha; o3[e] *= alpha;
      p0[e] = pa; p1[e] = pb;
    }

    // P (C layout) -> LDS -> A layout
    bf16* Pb = (bf16*)Pm;
#pragma unroll
    for (int e = 0; e < 8; ++e) {
      int r = e + 8 * g;
      Pb[r * 32 + nl]      = (bf16)p0[e];
      Pb[r * 32 + 16 + nl] = (bf16)p1[e];
    }
    wave_lds_fence();

    v16bf pf = frag_a(Pm, 16, lane);
    o0 = wmma_bf16(pf, frag_b(Vt, 16, lane), o0);
    o1 = wmma_bf16(pf, frag_b(Vt + 16 * 16, 16, lane), o1);
    o2 = wmma_bf16(pf, frag_b(Vt + 32 * 16, 16, lane), o2);
    o3 = wmma_bf16(pf, frag_b(Vt + 48 * 16, 16, lane), o3);
  }

  bf16* Ob = O + ((size_t)(bb * S_)) * QD_ + h * HD_;
#pragma unroll
  for (int e = 0; e < 8; ++e) {
    int row = q0 + e + 8 * g;
    float inv = 1.f / l_run[e];
    bf16* orow = Ob + (size_t)row * QD_;
    orow[nl]      = (bf16)(o0[e] * inv);
    orow[16 + nl] = (bf16)(o1[e] * inv);
    orow[32 + nl] = (bf16)(o2[e] * inv);
    orow[48 + nl] = (bf16)(o3[e] * inv);
  }
}

// ---------------------------------------------------------------------------
// Host launcher
// ---------------------------------------------------------------------------
namespace {
struct Alloc {
  char* base; size_t off;
  void* take(size_t bytes) {
    off = (off + 255) & ~(size_t)255;
    void* p = base + off;
    off += bytes;
    return p;
  }
};
inline void cast_launch(const void* in, void* out, int n, hipStream_t s) {
  cast_f32_bf16_kernel<<<(n + 255) / 256, 256, 0, s>>>(
      (const float*)in, (bf16*)out, n);
}
inline void gemm(const void* A, const void* Bw, const void* bias,
                 void* oF, void* oB, int M, int N, int K, int relu,
                 hipStream_t s) {
  dim3 grid(N / 64, M / 64);
  gemm_wmma_kernel<<<grid, 256, 0, s>>>(
      (const bf16*)A, (const bf16*)Bw, (const float*)bias,
      (float*)oF, (bf16*)oB, M, N, K, relu);
}
}  // namespace

extern "C" void kernel_launch(void* const* d_in, const int* in_sizes, int n_in,
                              void* d_out, int out_size, void* d_ws,
                              size_t ws_size, hipStream_t stream) {
  const void* hidden = d_in[0];
  const void* Wih_q = d_in[1]; const void* Whh_q = d_in[2]; const void* b_q = d_in[3];
  const void* Wih_k = d_in[4]; const void* Whh_k = d_in[5]; const void* b_k = d_in[6];
  const void* Wq1 = d_in[7];  const void* bq1 = d_in[8];
  const void* Wq2 = d_in[9];  const void* bq2 = d_in[10];
  const void* Wk1 = d_in[11]; const void* bk1 = d_in[12];
  const void* Wk2 = d_in[13]; const void* bk2 = d_in[14];
  const void* Wv  = d_in[15]; const void* bv  = d_in[16];
  const void* Wo  = d_in[17];

  Alloc ws{(char*)d_ws, 0};
  // bf16 casts
  void* hs_bf   = ws.take((size_t)MTOT_ * D_ * 2);
  void* Wihq_bf = ws.take((size_t)G4D_ * D_ * 2);
  void* Whhq_bf = ws.take((size_t)G4D_ * D_ * 2);
  void* Wihk_bf = ws.take((size_t)G4D_ * D_ * 2);
  void* Whhk_bf = ws.take((size_t)G4D_ * D_ * 2);
  void* Wq1_bf  = ws.take((size_t)OH_ * D_ * 2);
  void* Wq2_bf  = ws.take((size_t)QD_ * OH_ * 2);
  void* Wk1_bf  = ws.take((size_t)OH_ * D_ * 2);
  void* Wk2_bf  = ws.take((size_t)KD_ * OH_ * 2);
  void* Wv_bf   = ws.take((size_t)KD_ * D_ * 2);
  void* Wo_bf   = ws.take((size_t)D_ * QD_ * 2);
  // activations
  void* xWq   = ws.take((size_t)MTOT_ * G4D_ * 4);   // f32
  void* xWk   = ws.take((size_t)MTOT_ * G4D_ * 4);   // f32
  void* hq_bf = ws.take((size_t)MTOT_ * D_ * 2);
  void* hk_bf = ws.take((size_t)MTOT_ * D_ * 2);
  void* t1q   = ws.take((size_t)MTOT_ * OH_ * 2);
  void* t1k   = ws.take((size_t)MTOT_ * OH_ * 2);
  void* q_bf  = ws.take((size_t)MTOT_ * QD_ * 2);
  void* k_bf  = ws.take((size_t)MTOT_ * KD_ * 2);
  void* v_bf  = ws.take((size_t)MTOT_ * KD_ * 2);
  void* ao_bf = ws.take((size_t)MTOT_ * QD_ * 2);
  // LSTM scratch
  void* h_pad = ws.take((size_t)16 * D_ * 2);
  void* c_st  = ws.take((size_t)B_ * D_ * 4);
  void* gates = ws.take((size_t)B_ * G4D_ * 4);
  void* bar   = ws.take(256);

  (void)hipMemsetAsync(bar, 0, 256, stream);

  // casts
  cast_launch(hidden, hs_bf, MTOT_ * D_, stream);
  cast_launch(Wih_q, Wihq_bf, G4D_ * D_, stream);
  cast_launch(Whh_q, Whhq_bf, G4D_ * D_, stream);
  cast_launch(Wih_k, Wihk_bf, G4D_ * D_, stream);
  cast_launch(Whh_k, Whhk_bf, G4D_ * D_, stream);
  cast_launch(Wq1, Wq1_bf, OH_ * D_, stream);
  cast_launch(Wq2, Wq2_bf, QD_ * OH_, stream);
  cast_launch(Wk1, Wk1_bf, OH_ * D_, stream);
  cast_launch(Wk2, Wk2_bf, KD_ * OH_, stream);
  cast_launch(Wv, Wv_bf, KD_ * D_, stream);
  cast_launch(Wo, Wo_bf, D_ * QD_, stream);

  // input projections (bias folded here, matching xW = x@Wih.T + b)
  gemm(hs_bf, Wihq_bf, b_q, xWq, nullptr, MTOT_, G4D_, D_, 0, stream);
  gemm(hs_bf, Wihk_bf, b_k, xWk, nullptr, MTOT_, G4D_, D_, 0, stream);
  // value projection
  gemm(hs_bf, Wv_bf, bv, nullptr, v_bf, MTOT_, KD_, D_, 0, stream);

  // serial LSTMs (persistent, grid-synced)
  lstm_kernel<<<32, 256, 0, stream>>>((const float*)xWq, (const bf16*)Whhq_bf,
                                      (bf16*)hq_bf, (bf16*)h_pad, (float*)c_st,
                                      (float*)gates, (unsigned*)bar);
  lstm_kernel<<<32, 256, 0, stream>>>((const float*)xWk, (const bf16*)Whhk_bf,
                                      (bf16*)hk_bf, (bf16*)h_pad, (float*)c_st,
                                      (float*)gates, (unsigned*)bar);

  // out_q / out_k MLPs
  gemm(hq_bf, Wq1_bf, bq1, nullptr, t1q, MTOT_, OH_, D_, 1, stream);
  gemm(t1q, Wq2_bf, bq2, nullptr, q_bf, MTOT_, QD_, OH_, 0, stream);
  gemm(hk_bf, Wk1_bf, bk1, nullptr, t1k, MTOT_, OH_, D_, 1, stream);
  gemm(t1k, Wk2_bf, bk2, nullptr, k_bf, MTOT_, KD_, OH_, 0, stream);

  // causal GQA flash attention
  attn_kernel<<<dim3(S_ / 64, B_ * H_), 128, 0, stream>>>(
      (const bf16*)q_bf, (const bf16*)k_bf, (const bf16*)v_bf, (bf16*)ao_bf);

  // output projection -> d_out (f32)
  gemm(ao_bf, Wo_bf, nullptr, d_out, nullptr, MTOT_, D_, QD_, 0, stream);
}
